// M3GNet_74371653697920
// MI455X (gfx1250) — matrix operand order repaired and checked by reading, not compiled, not run on hardware
//
#include <hip/hip_runtime.h>
#include <hip/hip_bf16.h>
#include <math.h>

typedef __attribute__((ext_vector_type(16))) _Float16 v16h;
typedef __attribute__((ext_vector_type(8)))  _Float16 v8h;
typedef __attribute__((ext_vector_type(8)))  float    v8f;

#define HID 64

__device__ __forceinline__ float sigmoidf_(float x) { return 1.0f / (1.0f + expf(-x)); }
__device__ __forceinline__ float siluf_(float x)    { return x * sigmoidf_(x); }

// ----------------------------------------------------------------------------
// WMMA fragment builders.
// A layout (16-bit 16xK): lanes 0-15 -> M=lane, lanes 16-31 -> M=lane-16;
//   element e -> K = k0 + e + (half?16:0)
//   => one contiguous 16-wide K run: [b1, b1+15], b1=k0+16*half
// B layout (16-bit Kx16): lane&15 -> N; element e -> K = k0 + e + (half?16:0)
//   => 16 contiguous K per lane -> one v16h load from transposed f16 weights
// C/D layout: vgpr v -> M = v + 8*half; lane&15 -> N
// ----------------------------------------------------------------------------
template <class FPA>
__device__ __forceinline__ v16h make_a_g(FPA aptr, int k0, int hi) {
  const int b1 = k0 + (hi ? 16 : 0);
  const float4* p0 = (const float4*)aptr(b1);
  float4 u0 = p0[0], u1 = p0[1], w0 = p0[2], w1 = p0[3];
  v16h a;
  a[0] = (_Float16)u0.x; a[1] = (_Float16)u0.y; a[2]  = (_Float16)u0.z; a[3]  = (_Float16)u0.w;
  a[4] = (_Float16)u1.x; a[5] = (_Float16)u1.y; a[6]  = (_Float16)u1.z; a[7]  = (_Float16)u1.w;
  a[8] = (_Float16)w0.x; a[9] = (_Float16)w0.y; a[10] = (_Float16)w0.z; a[11] = (_Float16)w0.w;
  a[12] = (_Float16)w1.x; a[13] = (_Float16)w1.y; a[14] = (_Float16)w1.z; a[15] = (_Float16)w1.w;
  return a;
}

__device__ __forceinline__ v16h make_a_lds(const _Float16* arow, int k0, int hi) {
  const int b1 = k0 + (hi ? 16 : 0);
  v8h r0 = *(const v8h*)(arow + b1);
  v8h r1 = *(const v8h*)(arow + b1 + 8);
  v16h a;
#pragma unroll
  for (int i = 0; i < 8; ++i) { a[i] = r0[i]; a[8 + i] = r1[i]; }
  return a;
}

// Global-A, two transposed-f16 B matrices (shared A gather for m/g paths).
template <int K, class FPA>
__device__ __forceinline__ void wmma_dual_t(FPA aptr,
                                            const _Float16* __restrict__ WtM,
                                            const _Float16* __restrict__ WtG,
                                            v8f accm[4], v8f accg[4]) {
  const int lane = threadIdx.x & 31;
  const int hi = lane >> 4, ln = lane & 15;
  for (int k0 = 0; k0 < K; k0 += 32) {
    v16h a = make_a_g(aptr, k0, hi);
    const int kb = k0 + (hi ? 16 : 0);
#pragma unroll
    for (int nt = 0; nt < 4; ++nt) {
      const int n = nt * 16 + ln;
      v16h bm = *(const v16h*)(WtM + (size_t)n * K + kb);
      v16h bg = *(const v16h*)(WtG + (size_t)n * K + kb);
      accm[nt] = __builtin_amdgcn_wmma_f32_16x16x32_f16(
          false, a, false, bm, (short)0, accm[nt], false, false);
      accg[nt] = __builtin_amdgcn_wmma_f32_16x16x32_f16(
          false, a, false, bg, (short)0, accg[nt], false, false);
    }
  }
}

// LDS-A (f16 row per lane), one transposed-f16 B matrix.
template <int K>
__device__ __forceinline__ void wmma_lds_t(const _Float16* arow,
                                           const _Float16* __restrict__ Wt,
                                           v8f acc[4]) {
  const int lane = threadIdx.x & 31;
  const int hi = lane >> 4, ln = lane & 15;
  for (int k0 = 0; k0 < K; k0 += 32) {
    v16h a = make_a_lds(arow, k0, hi);
    const int kb = k0 + (hi ? 16 : 0);
#pragma unroll
    for (int nt = 0; nt < 4; ++nt) {
      const int n = nt * 16 + ln;
      v16h b = *(const v16h*)(Wt + (size_t)n * K + kb);
      acc[nt] = __builtin_amdgcn_wmma_f32_16x16x32_f16(
          false, a, false, b, (short)0, acc[nt], false, false);
    }
  }
}

__device__ __forceinline__ void zero_acc(v8f acc[4]) {
#pragma unroll
  for (int nt = 0; nt < 4; ++nt)
#pragma unroll
    for (int i = 0; i < 8; ++i) acc[nt][i] = 0.0f;
}

// ----------------------------------------------------------------------------
// Weight transpose+convert: W[l][k][n] (f32) -> Wt[l][n][k] (f16)
// ----------------------------------------------------------------------------
__global__ void k_transp(const float* __restrict__ W, _Float16* __restrict__ Wt,
                         int L, int K, int Nn) {
  int idx = blockIdx.x * blockDim.x + threadIdx.x;
  if (idx >= L * K * Nn) return;
  int l = idx / (K * Nn);
  int rem = idx - l * K * Nn;
  int k = rem / Nn, n = rem - k * Nn;
  Wt[((size_t)l * Nn + n) * K + k] = (_Float16)W[idx];
}

// ----------------------------------------------------------------------------
// Small precompute kernels
// ----------------------------------------------------------------------------
__global__ void k_atom_init(const float* __restrict__ embW, const int* __restrict__ z,
                            float* __restrict__ atom, int N) {
  int idx = blockIdx.x * blockDim.x + threadIdx.x;
  if (idx >= N * HID) return;
  int n = idx >> 6, h = idx & 63;
  atom[idx] = embW[z[n] * HID + h];
}

__global__ void k_rbf(const float* __restrict__ elen, float* __restrict__ rbf0, int E) {
  int e = blockIdx.x * blockDim.x + threadIdx.x;
  if (e >= E) return;
  const float PI = 3.14159265358979323846f;
  const float C = 5.0f;
  float r = elen[e];
  float en[4], dn[4], fnr[4];
#pragma unroll
  for (int n = 0; n < 4; ++n) {
    float fn = (float)n;
    float p = (fn + 1.0f) * (fn + 1.0f) * (fn + 1.0f) * (fn + 1.0f);
    en[n] = fn * fn * (fn + 2.0f) * (fn + 2.0f) / (4.0f * p + 1.0f);
  }
  dn[0] = 1.0f;
#pragma unroll
  for (int i = 1; i < 4; ++i) dn[i] = 1.0f - en[i] / dn[i - 1];
#pragma unroll
  for (int n = 0; n < 4; ++n) {
    float fn = (float)n;
    float coef = ((n & 1) ? -1.0f : 1.0f) * sqrtf(2.0f) * PI / (C * sqrtf(C)) *
                 (fn + 1.0f) * (fn + 2.0f) /
                 sqrtf((fn + 1.0f) * (fn + 1.0f) + (fn + 2.0f) * (fn + 2.0f));
    float x1 = r * (fn + 1.0f) * PI / C;
    float x2 = r * (fn + 2.0f) * PI / C;
    fnr[n] = coef * (sinf(x1) / x1 + sinf(x2) / x2);
  }
  float g = fnr[0];
  rbf0[(size_t)e * 4 + 0] = g;
#pragma unroll
  for (int i = 1; i < 4; ++i) {
    g = (fnr[i] + sqrtf(en[i] / dn[i - 1]) * g) / sqrtf(dn[i]);
    rbf0[(size_t)e * 4 + i] = g;
  }
}

__global__ void k_edge_init(const float* __restrict__ rbf0, const float* __restrict__ W,
                            float* __restrict__ edge, int E) {
  int idx = blockIdx.x * blockDim.x + threadIdx.x;
  if (idx >= E * HID) return;
  int e = idx >> 6, h = idx & 63;
  float acc = 0.0f;
#pragma unroll
  for (int n = 0; n < 4; ++n) acc += rbf0[(size_t)e * 4 + n] * W[n * HID + h];
  edge[idx] = siluf_(acc);
}

__device__ const float SBR[4][4] = {
    {3.14159265f, 6.28318531f, 9.42477796f, 12.56637061f},
    {4.49340946f, 7.72525184f, 10.90412166f, 14.06619391f},
    {5.76345920f, 9.09501133f, 12.32294097f, 15.51460301f},
    {6.98793200f, 10.41711855f, 13.69802315f, 16.92362129f}};

__device__ __forceinline__ float sph_jl(int l, float x) {
  float s = sinf(x), c = cosf(x);
  float x2 = x * x, x3 = x2 * x, x4 = x3 * x, x5 = x4 * x;
  if (l == 0) return s / x;
  if (l == 1) return s / x2 - c / x;
  if (l == 2) return (3.0f / x3 - 1.0f / x) * s - 3.0f / x2 * c;
  if (l == 3) return (15.0f / x4 - 6.0f / x2) * s - (15.0f / x3 - 1.0f / x) * c;
  return (105.0f / x5 - 45.0f / x3 + 1.0f / x) * s - (105.0f / x4 - 10.0f / x2) * c;
}

__device__ __forceinline__ float cutf(float r, float c) {
  float x = r / c;
  float x3 = x * x * x;
  return (r <= c) ? (1.0f - 6.0f * x3 * x * x + 15.0f * x3 * x - 10.0f * x3) : 0.0f;
}

__global__ void k_triples(const float* __restrict__ evec, const float* __restrict__ elen,
                          const int* __restrict__ tb0, const int* __restrict__ tb1,
                          float* __restrict__ sbf, int T) {
  int t = blockIdx.x * blockDim.x + threadIdx.x;
  if (t >= T) return;
  int e0 = tb0[t], e1 = tb1[t];
  float a0 = evec[(size_t)e0 * 3 + 0], a1 = evec[(size_t)e0 * 3 + 1], a2 = evec[(size_t)e0 * 3 + 2];
  float b0 = evec[(size_t)e1 * 3 + 0], b1 = evec[(size_t)e1 * 3 + 1], b2 = evec[(size_t)e1 * 3 + 2];
  float rij = elen[e0], rik = elen[e1];
  float ct = (a0 * b0 + a1 * b1 + a2 * b2) / (rij * rik);
  ct = fminf(fmaxf(ct, -1.0f + 1e-7f), 1.0f - 1e-7f);
  float fc = cutf(rij, 4.0f) * cutf(rik, 4.0f);
  float P[4];
  P[0] = 1.0f;
  P[1] = ct;
  P[2] = 0.5f * (3.0f * ct * ct - 1.0f);
  P[3] = 0.5f * (5.0f * ct * ct * ct - 3.0f * ct);
  const float FOURPI = 12.5663706144f;
  const float FAC = 0.126491106407f;  // sqrt(2/125)
  float* o = sbf + (size_t)t * 16;
#pragma unroll
  for (int l = 0; l < 4; ++l) {
    float y = sqrtf((2.0f * l + 1.0f) / FOURPI) * P[l];
#pragma unroll
    for (int n = 0; n < 4; ++n) {
      float root = SBR[l][n];
      float x = rik * root / 5.0f;
      float jl = sph_jl(l, x);
      float nrm = 1.0f / fabsf(sph_jl(l + 1, root));
      o[l * 4 + n] = FAC * nrm * jl * y * fc;
    }
  }
}

// ----------------------------------------------------------------------------
// Per-layer scalar kernels
// ----------------------------------------------------------------------------
__global__ void k_sig(const float* __restrict__ atom, const float* __restrict__ W,
                      const float* __restrict__ b, float* __restrict__ sig, int N) {
  int idx = blockIdx.x * blockDim.x + threadIdx.x;
  if (idx >= N * 16) return;
  int n = idx >> 4, j = idx & 15;
  const float* x = atom + (size_t)n * HID;
  float acc = b[j];
  for (int h = 0; h < HID; ++h) acc += x[h] * W[h * 16 + j];
  sig[idx] = sigmoidf_(acc);
}

__global__ void k_zero(float* __restrict__ p, int cnt) {
  int idx = blockIdx.x * blockDim.x + threadIdx.x;
  if (idx < cnt) p[idx] = 0.0f;
}

__global__ void k_copy(const float* __restrict__ s, float* __restrict__ d, int cnt) {
  int idx = blockIdx.x * blockDim.x + threadIdx.x;
  if (idx < cnt) d[idx] = s[idx];
}

__global__ void k_scatter(const float* __restrict__ sbf, const float* __restrict__ sig,
                          const int* __restrict__ tb0, const int* __restrict__ tb1,
                          const int* __restrict__ idx_t, float* __restrict__ eprime, int T) {
  int t = blockIdx.x * blockDim.x + threadIdx.x;
  if (t >= T) return;
  int e0 = tb0[t];
  int a = idx_t[tb1[t]];
  const float* sb = sbf + (size_t)t * 16;
  const float* sg = sig + (size_t)a * 16;
  float* dst = eprime + (size_t)e0 * 16;
#pragma unroll
  for (int j = 0; j < 16; ++j) atomicAdd(dst + j, sb[j] * sg[j]);
}

__global__ void k_tbup(const float* __restrict__ eprime, const float* __restrict__ mW,
                       const float* __restrict__ gW, float* __restrict__ edge, int E) {
  int idx = blockIdx.x * blockDim.x + threadIdx.x;
  if (idx >= E * HID) return;
  int e = idx >> 6, h = idx & 63;
  const float* ep = eprime + (size_t)e * 16;
  float m = 0.0f, g = 0.0f;
#pragma unroll
  for (int j = 0; j < 16; ++j) {
    m += ep[j] * mW[j * HID + h];
    g += ep[j] * gW[j * HID + h];
  }
  edge[idx] += siluf_(m) * sigmoidf_(g);
}

// ----------------------------------------------------------------------------
// Gated2 update (edge / atom), WMMA. One wave = 16-edge tile.
// feat = [atom[s], atom[t], edge] : 16x192 ; stage1 -> 16x64 (silu) ;
// LDS relayout ; stage2 -> 16x64 ; fused epilogue with rbf gate.
// ----------------------------------------------------------------------------
template <bool ATOM>
__global__ void __launch_bounds__(256) gated2_kernel(
    const float* __restrict__ atom_cur, float* __restrict__ edge_attr,
    float* __restrict__ atom_next, const int* __restrict__ idx_s,
    const int* __restrict__ idx_t, const float* __restrict__ rbf0,
    const _Float16* __restrict__ Wt1m, const float* __restrict__ b1m,
    const _Float16* __restrict__ Wt2m, const float* __restrict__ b2m,
    const _Float16* __restrict__ Wt1g, const float* __restrict__ b1g,
    const _Float16* __restrict__ Wt2g, const float* __restrict__ b2g,
    const float* __restrict__ rbfW, int nEdges) {
  __shared__ _Float16 sh[8][2][16][HID];
  const int wave = threadIdx.x >> 5;
  const int lane = threadIdx.x & 31;
  const int hi = lane >> 4, ln = lane & 15;
  const int tile = blockIdx.x * 8 + wave;
  if (tile * 16 >= nEdges) return;

  const int erow = tile * 16 + ln;
  const float* sp = atom_cur + (size_t)idx_s[erow] * HID;
  const float* tp = atom_cur + (size_t)idx_t[erow] * HID;
  const float* epr = edge_attr + (size_t)erow * HID;

  v8f accm[4], accg[4];
  zero_acc(accm);
  zero_acc(accg);

  // ---- stage 1: feat(16x192) @ W1(192x64), shared A for m/g.
  // Each 16-wide K run stays inside one 64-wide feat section (k0 % 32 == 0).
  wmma_dual_t<192>(
      [&](int b) -> const float* {
        return (b < 64) ? (sp + b) : ((b < 128) ? (tp + b - 64) : (epr + b - 128));
      },
      Wt1m, Wt1g, accm, accg);

#pragma unroll
  for (int nt = 0; nt < 4; ++nt) {
    int Nc = nt * 16 + ln;
    float bm = b1m[Nc], bg = b1g[Nc];
#pragma unroll
    for (int v = 0; v < 8; ++v) {
      int M = v + hi * 8;
      sh[wave][0][M][Nc] = (_Float16)siluf_(accm[nt][v] + bm);
      sh[wave][1][M][Nc] = (_Float16)siluf_(accg[nt][v] + bg);
    }
  }

  // ---- stage 2: h(16x64) @ W2(64x64)
  zero_acc(accm);
  zero_acc(accg);
  wmma_lds_t<64>(&sh[wave][0][ln][0], Wt2m, accm);
  wmma_lds_t<64>(&sh[wave][1][ln][0], Wt2g, accg);

  // ---- epilogue: silu(m) * sigmoid(g) * (rbf0 @ rbfW); add or scatter
#pragma unroll
  for (int nt = 0; nt < 4; ++nt) {
    int Nc = nt * 16 + ln;
    float bm = b2m[Nc], bg = b2g[Nc];
    float rw0 = rbfW[0 * HID + Nc], rw1 = rbfW[1 * HID + Nc];
    float rw2 = rbfW[2 * HID + Nc], rw3 = rbfW[3 * HID + Nc];
#pragma unroll
    for (int v = 0; v < 8; ++v) {
      int M = v + hi * 8;
      int er = tile * 16 + M;
      float m = siluf_(accm[nt][v] + bm);
      float g = sigmoidf_(accg[nt][v] + bg);
      const float* rb = rbf0 + (size_t)er * 4;
      float r = rb[0] * rw0 + rb[1] * rw1 + rb[2] * rw2 + rb[3] * rw3;
      float val = m * g * r;
      if (ATOM) {
        atomicAdd(&atom_next[(size_t)idx_s[er] * HID + Nc], val);
      } else {
        edge_attr[(size_t)er * HID + Nc] += val;
      }
    }
  }
}

// ----------------------------------------------------------------------------
// Gated3 head (energy / forces): 3-stage m-chain & g-chain, WMMA + LDS staging
// Wt layouts: [stage][n][k] f16, stage stride 4096.
// ----------------------------------------------------------------------------
__global__ void __launch_bounds__(256) gated3_kernel(
    const float* __restrict__ X, int R,
    const _Float16* __restrict__ Wtm, const float* __restrict__ mb,
    const _Float16* __restrict__ Wtg, const float* __restrict__ gb,
    float* __restrict__ out) {
  __shared__ _Float16 sh[8][2][16][HID];
  const int wave = threadIdx.x >> 5;
  const int lane = threadIdx.x & 31;
  const int hi = lane >> 4, ln = lane & 15;
  const int tile = blockIdx.x * 8 + wave;
  if (tile * 16 >= R) return;

  const float* xr = X + ((size_t)tile * 16 + ln) * HID;
  v8f am[4], ag[4];
  zero_acc(am);
  zero_acc(ag);

  // stage 0 (shared A = input rows)
  wmma_dual_t<64>([&](int b) -> const float* { return xr + b; }, Wtm, Wtg, am, ag);
#pragma unroll
  for (int nt = 0; nt < 4; ++nt) {
    int Nc = nt * 16 + ln;
    float bm = mb[Nc], bg = gb[Nc];
#pragma unroll
    for (int v = 0; v < 8; ++v) {
      int M = v + hi * 8;
      sh[wave][0][M][Nc] = (_Float16)siluf_(am[nt][v] + bm);
      sh[wave][1][M][Nc] = (_Float16)siluf_(ag[nt][v] + bg);
    }
  }

  // stage 1
  zero_acc(am);
  zero_acc(ag);
  wmma_lds_t<64>(&sh[wave][0][ln][0], Wtm + 4096, am);
  wmma_lds_t<64>(&sh[wave][1][ln][0], Wtg + 4096, ag);
#pragma unroll
  for (int nt = 0; nt < 4; ++nt) {
    int Nc = nt * 16 + ln;
    float bm = mb[64 + Nc], bg = gb[64 + Nc];
#pragma unroll
    for (int v = 0; v < 8; ++v) {
      int M = v + hi * 8;
      sh[wave][0][M][Nc] = (_Float16)siluf_(am[nt][v] + bm);
      sh[wave][1][M][Nc] = (_Float16)siluf_(ag[nt][v] + bg);
    }
  }

  // stage 2: m linear, g sigmoid; out = m * g
  zero_acc(am);
  zero_acc(ag);
  wmma_lds_t<64>(&sh[wave][0][ln][0], Wtm + 8192, am);
  wmma_lds_t<64>(&sh[wave][1][ln][0], Wtg + 8192, ag);
#pragma unroll
  for (int nt = 0; nt < 4; ++nt) {
    int Nc = nt * 16 + ln;
    float bm = mb[128 + Nc], bg = gb[128 + Nc];
#pragma unroll
    for (int v = 0; v < 8; ++v) {
      int M = v + hi * 8;
      float m = am[nt][v] + bm;
      float g = sigmoidf_(ag[nt][v] + bg);
      out[((size_t)tile * 16 + M) * HID + Nc] = m * g;
    }
  }
}

// ----------------------------------------------------------------------------
// Launcher
// ----------------------------------------------------------------------------
extern "C" void kernel_launch(void* const* d_in, const int* in_sizes, int n_in,
                              void* d_out, int out_size, void* d_ws, size_t ws_size,
                              hipStream_t stream) {
  (void)n_in; (void)out_size; (void)ws_size;
  // Inputs: params leaves (dict order) then graph arrays.
  // 0 atom_emb_W, 1 edge_enc_W, 2 tb_sig_W, 3 tb_sig_b, 4 tb_mW, 5 tb_gW,
  // 6 eu_mW1, 7 eu_mb1, 8 eu_mW2, 9 eu_mb2, 10 eu_gW1, 11 eu_gb1, 12 eu_gW2,
  // 13 eu_gb2, 14 eu_rbf_W, 15..23 au_* (same pattern), 24 en_mW, 25 en_mb,
  // 26 en_gW, 27 en_gb, 28 fo_mW, 29 fo_mb, 30 fo_gW, 31 fo_gb,
  // 32 atomic_numbers, 33 edge_index, 34 edge_length, 35 edge_vector,
  // 36 three_body_indices, 37 triple_edge_length (unused)
  const float* P[32];
  for (int i = 0; i < 32; ++i) P[i] = (const float*)d_in[i];
  const int*   zN   = (const int*)d_in[32];
  const int*   ei   = (const int*)d_in[33];
  const float* elen = (const float*)d_in[34];
  const float* evec = (const float*)d_in[35];
  const int*   tbi  = (const int*)d_in[36];

  const int N = in_sizes[32];
  const int E = in_sizes[34];
  const int T = in_sizes[37];

  const int* idx_s = ei;
  const int* idx_t = ei + E;
  const int* tb0 = tbi;
  const int* tb1 = tbi + T;

  float* out = (float*)d_out;
  float* atomA  = out;                         // [N,64]
  float* edgeA  = atomA + (size_t)N * HID;     // [E,64]
  float* energy = edgeA + (size_t)E * HID;     // [N,64]
  float* forces = energy + (size_t)N * HID;    // [E,64]

  char* wsb = (char*)d_ws;
  size_t off = 0;
  auto walloc = [&](size_t bytes) {
    void* p = wsb + off;
    off = (off + bytes + 255) & ~(size_t)255;
    return p;
  };
  float* atomB  = (float*)walloc((size_t)N * HID * 4);
  float* rbf0   = (float*)walloc((size_t)E * 4 * 4);
  float* sbf    = (float*)walloc((size_t)T * 16 * 4);
  float* sig    = (float*)walloc((size_t)N * 16 * 4);
  float* eprime = (float*)walloc((size_t)E * 16 * 4);
  // transposed f16 weights
  _Float16* t_eu_m1 = (_Float16*)walloc((size_t)4 * 64 * 192 * 2);
  _Float16* t_eu_g1 = (_Float16*)walloc((size_t)4 * 64 * 192 * 2);
  _Float16* t_au_m1 = (_Float16*)walloc((size_t)4 * 64 * 192 * 2);
  _Float16* t_au_g1 = (_Float16*)walloc((size_t)4 * 64 * 192 * 2);
  _Float16* t_eu_m2 = (_Float16*)walloc((size_t)4 * 64 * 64 * 2);
  _Float16* t_eu_g2 = (_Float16*)walloc((size_t)4 * 64 * 64 * 2);
  _Float16* t_au_m2 = (_Float16*)walloc((size_t)4 * 64 * 64 * 2);
  _Float16* t_au_g2 = (_Float16*)walloc((size_t)4 * 64 * 64 * 2);
  _Float16* t_en_m  = (_Float16*)walloc((size_t)3 * 64 * 64 * 2);
  _Float16* t_en_g  = (_Float16*)walloc((size_t)3 * 64 * 64 * 2);
  _Float16* t_fo_m  = (_Float16*)walloc((size_t)3 * 64 * 64 * 2);
  _Float16* t_fo_g  = (_Float16*)walloc((size_t)3 * 64 * 64 * 2);

  const int thr = 256;
  auto nblk = [&](long long c) { return (unsigned)((c + thr - 1) / thr); };

  // weight transposes (tiny)
  k_transp<<<nblk(4LL * 192 * 64), thr, 0, stream>>>(P[6],  t_eu_m1, 4, 192, 64);
  k_transp<<<nblk(4LL * 192 * 64), thr, 0, stream>>>(P[10], t_eu_g1, 4, 192, 64);
  k_transp<<<nblk(4LL * 192 * 64), thr, 0, stream>>>(P[15], t_au_m1, 4, 192, 64);
  k_transp<<<nblk(4LL * 192 * 64), thr, 0, stream>>>(P[19], t_au_g1, 4, 192, 64);
  k_transp<<<nblk(4LL * 64 * 64),  thr, 0, stream>>>(P[8],  t_eu_m2, 4, 64, 64);
  k_transp<<<nblk(4LL * 64 * 64),  thr, 0, stream>>>(P[12], t_eu_g2, 4, 64, 64);
  k_transp<<<nblk(4LL * 64 * 64),  thr, 0, stream>>>(P[17], t_au_m2, 4, 64, 64);
  k_transp<<<nblk(4LL * 64 * 64),  thr, 0, stream>>>(P[21], t_au_g2, 4, 64, 64);
  k_transp<<<nblk(3LL * 64 * 64),  thr, 0, stream>>>(P[24], t_en_m, 3, 64, 64);
  k_transp<<<nblk(3LL * 64 * 64),  thr, 0, stream>>>(P[26], t_en_g, 3, 64, 64);
  k_transp<<<nblk(3LL * 64 * 64),  thr, 0, stream>>>(P[28], t_fo_m, 3, 64, 64);
  k_transp<<<nblk(3LL * 64 * 64),  thr, 0, stream>>>(P[30], t_fo_g, 3, 64, 64);

  k_atom_init<<<nblk((long long)N * HID), thr, 0, stream>>>(P[0], zN, atomA, N);
  k_rbf<<<nblk(E), thr, 0, stream>>>(elen, rbf0, E);
  k_edge_init<<<nblk((long long)E * HID), thr, 0, stream>>>(rbf0, P[1], edgeA, E);
  k_triples<<<nblk(T), thr, 0, stream>>>(evec, elen, tb0, tb1, sbf, T);

  const int tiles = E / 16;
  const unsigned g2blk = (unsigned)((tiles + 7) / 8);

  float* cur = atomA;
  float* nxt = atomB;
  for (int l = 0; l < 4; ++l) {
    k_sig<<<nblk((long long)N * 16), thr, 0, stream>>>(
        cur, P[2] + (size_t)l * HID * 16, P[3] + (size_t)l * 16, sig, N);
    k_zero<<<nblk((long long)E * 16), thr, 0, stream>>>(eprime, E * 16);
    k_scatter<<<nblk(T), thr, 0, stream>>>(sbf, sig, tb0, tb1, idx_t, eprime, T);
    k_tbup<<<nblk((long long)E * HID), thr, 0, stream>>>(
        eprime, P[4] + (size_t)l * 16 * HID, P[5] + (size_t)l * 16 * HID, edgeA, E);

    // edge (bond) update
    gated2_kernel<false><<<g2blk, 256, 0, stream>>>(
        cur, edgeA, nullptr, idx_s, idx_t, rbf0,
        t_eu_m1 + (size_t)l * 64 * 192, P[7]  + (size_t)l * HID,
        t_eu_m2 + (size_t)l * 64 * 64,  P[9]  + (size_t)l * HID,
        t_eu_g1 + (size_t)l * 64 * 192, P[11] + (size_t)l * HID,
        t_eu_g2 + (size_t)l * 64 * 64,  P[13] + (size_t)l * HID,
        P[14] + (size_t)l * 4 * HID, E);

    // atom update: copy cur -> nxt, then atomic scatter-add into nxt
    k_copy<<<nblk((long long)N * HID), thr, 0, stream>>>(cur, nxt, N * HID);
    gated2_kernel<true><<<g2blk, 256, 0, stream>>>(
        cur, edgeA, nxt, idx_s, idx_t, rbf0,
        t_au_m1 + (size_t)l * 64 * 192, P[16] + (size_t)l * HID,
        t_au_m2 + (size_t)l * 64 * 64,  P[18] + (size_t)l * HID,
        t_au_g1 + (size_t)l * 64 * 192, P[20] + (size_t)l * HID,
        t_au_g2 + (size_t)l * 64 * 64,  P[22] + (size_t)l * HID,
        P[23] + (size_t)l * 4 * HID, E);

    float* tmp = cur; cur = nxt; nxt = tmp;  // after 4 swaps, cur == atomA
  }

  gated3_kernel<<<(unsigned)((N / 16 + 7) / 8), 256, 0, stream>>>(
      cur, N, t_en_m, P[25], t_en_g, P[27], energy);
  gated3_kernel<<<(unsigned)((E / 16 + 7) / 8), 256, 0, stream>>>(
      edgeA, E, t_fo_m, P[29], t_fo_g, P[31], forces);
}
